// joint_test_38732015075541
// MI455X (gfx1250) — matrix operand order repaired and checked by reading, not compiled, not run on hardware
//
#include <hip/hip_runtime.h>

typedef __attribute__((ext_vector_type(16))) __bf16 v16bf;
typedef __attribute__((ext_vector_type(8)))  float  v8f;

#define T_DIM  8192
#define DIME_C 8
#define A3_C   256
#define CWIN   16
#define COUNT_C 4
#define HDIM   128
#define PL_C   24
#define BN_INVC 0.9999950000374996f  // 1/sqrt(1+1e-5)

// ---------------------------------------------------------------------------
// Pre-pack a weight matrix into WMMA B-fragment order (bf16).
// B logical matrix is K x N (K = Ktiles*32, N = Ntiles*16).
// B[k][n] = transB ? W[n*K + k] : W[k*N + n]
// Fragment layout (16-bit B, 16x16x32): lane = h*16 + ln, element i:
//   k = ktile*32 + ((i<8) ? h*8+i : 16+h*8+(i-8)) ; n = ntile*16 + ln
// dst index = ((ntile*Ktiles + ktile)*32 + lane)*16 + i
// ---------------------------------------------------------------------------
__global__ void swizzleB_kernel(const float* __restrict__ W, __bf16* __restrict__ dst,
                                int Ntiles, int Ktiles, int N, int K, int transB,
                                long srcBatchStride, long dstBatchStride) {
  const float* Wb = W + (long)blockIdx.y * srcBatchStride;
  __bf16* db = dst + (long)blockIdx.y * dstBatchStride;
  int idx = blockIdx.x * blockDim.x + threadIdx.x;
  int total = Ntiles * Ktiles * 512;
  if (idx >= total) return;
  int i    = idx & 15;
  int lane = (idx >> 4) & 31;
  int kt   = (idx >> 9) % Ktiles;
  int nt   = (idx >> 9) / Ktiles;
  int h = lane >> 4, ln = lane & 15;
  int k = kt * 32 + ((i < 8) ? (h * 8 + i) : (16 + h * 8 + (i - 8)));
  int n = nt * 16 + ln;
  float v = transB ? Wb[(long)n * K + k] : Wb[(long)k * N + n];
  db[idx] = (__bf16)v;
}

// ---------------------------------------------------------------------------
// Stage 1: x[t,a] = sum_{v,d} x_mul[v,t,d,a] * merge_w[v*8+d, a]
// Dominant memory stream (256 MB): float4 so every access is b128.
// ---------------------------------------------------------------------------
__global__ void merge_kernel(const float* __restrict__ xmul, const float* __restrict__ mw,
                             float* __restrict__ x) {
  long idx4 = ((long)blockIdx.x * 256 + threadIdx.x) * 4;  // t*A3 + a, a multiple of 4
  int a = (int)(idx4 & 255);
  long t = idx4 >> 8;
  float4 s = make_float4(0.f, 0.f, 0.f, 0.f);
  #pragma unroll
  for (int v = 0; v < COUNT_C; ++v) {
    #pragma unroll
    for (int d = 0; d < DIME_C; ++d) {
      const float4 xv = *(const float4*)&xmul[(((long)v * T_DIM + t) * DIME_C + d) * A3_C + a];
      const float4 wv = *(const float4*)&mw[(v * DIME_C + d) * A3_C + a];
      s.x += xv.x * wv.x; s.y += xv.y * wv.y;
      s.z += xv.z * wv.z; s.w += xv.w * wv.w;
    }
  }
  *(float4*)&x[idx4] = s;
}

// ---------------------------------------------------------------------------
// Stage 2: sliding window + L2-normalize. Writes rep (f32 output) and the
// A-fragment-swizzled bf16 copy of xr for the jiang GEMM.
// A layout (16-bit A 16x32): lane = h*16 + r; element i -> kk = (i<8)? h*8+i
//                            : 16 + h*8 + (i-8). Inverse used here for stores.
// ---------------------------------------------------------------------------
__global__ void window_norm_kernel(const float* __restrict__ x,
                                   const float* __restrict__ lw, const float* __restrict__ lb,
                                   float* __restrict__ rep_out, __bf16* __restrict__ afrag) {
  __shared__ float red[256];
  int t = blockIdx.x, a = threadIdx.x;
  float w[CWIN];
  #pragma unroll
  for (int k = 0; k < CWIN; ++k) w[k] = lw[k * A3_C + a];
  float bs = 0.f;
  #pragma unroll
  for (int k = 0; k < CWIN; ++k) bs += lb[k * A3_C + a];
  float s = bs;
  if (t >= CWIN - 1) {
    #pragma unroll
    for (int k = 0; k < CWIN; ++k) s += w[k] * x[(long)(t - 15 + k) * A3_C + a];
  } else {
    float tw = 0.f;
    for (int k = 0; k <= t; ++k) s += w[k] * x[(long)k * A3_C + a];
    for (int k = t + 1; k < CWIN; ++k) tw += w[k];
    s += tw * x[(long)t * A3_C + a];
  }
  red[a] = s * s;
  __syncthreads();
  for (int off = 128; off > 0; off >>= 1) {
    if (a < off) red[a] += red[a + off];
    __syncthreads();
  }
  float nrm = fmaxf(sqrtf(red[0]), 1e-12f);
  float v = s / nrm;
  rep_out[(long)t * A3_C + a] = v;
  // swizzled A-fragment store
  int kt = a >> 5, kk = a & 31;
  int h2 = (kk >> 3) & 1;
  int i2 = (kk & 7) | ((kk & 16) >> 1);
  int lane_t = (t & 15) + 16 * h2;
  afrag[(((long)(t >> 4) * 8 + kt) * 32 + lane_t) * 16 + i2] = (__bf16)v;
}

// ---------------------------------------------------------------------------
// Stage 3: per-channel jiang MLP.  Block = 16 rows x one d, 8 waves.
// GEMM xr(16x256) @ jW1[d](256x256) via bf16 WMMA, fused BN/ELU/jW2 reduce.
// ---------------------------------------------------------------------------
__global__ void jiang_kernel(const __bf16* __restrict__ afrag, const __bf16* __restrict__ bfragJ,
                             const float* __restrict__ jb1, const float* __restrict__ jgamma,
                             const float* __restrict__ jbeta, const float* __restrict__ jW2,
                             const float* __restrict__ jb2, float* __restrict__ xfina) {
  __shared__ float hbuf[16][A3_C];
  __shared__ float rbuf[16][16];
  int tt = blockIdx.x;   // row tile 0..511
  int d  = blockIdx.y;   // 0..7
  int lane = threadIdx.x & 31, wv = threadIdx.x >> 5;
  v8f acc0 = {0.f,0.f,0.f,0.f,0.f,0.f,0.f,0.f};
  v8f acc1 = acc0;
  const __bf16* bbase = bfragJ + (long)d * 16 * 8 * 512;
  int nt0 = wv * 2, nt1 = wv * 2 + 1;
  #pragma unroll
  for (int kt = 0; kt < 8; ++kt) {
    v16bf av = *(const v16bf*)(afrag + (((long)tt * 8 + kt) * 32 + lane) * 16);
    v16bf b0 = *(const v16bf*)(bbase + (((long)nt0 * 8 + kt) * 32 + lane) * 16);
    v16bf b1 = *(const v16bf*)(bbase + (((long)nt1 * 8 + kt) * 32 + lane) * 16);
    acc0 = __builtin_amdgcn_wmma_f32_16x16x32_bf16(false, av, false, b0, (short)0, acc0, false, false);
    acc1 = __builtin_amdgcn_wmma_f32_16x16x32_bf16(false, av, false, b1, (short)0, acc1, false, false);
  }
  int half = lane >> 4, ln = lane & 15;
  #pragma unroll
  for (int nt = 0; nt < 2; ++nt) {
    int col = (wv * 2 + nt) * 16 + ln;
    float g_ = jgamma[d * A3_C + col], be = jbeta[d * A3_C + col];
    float bb = jb1[d * A3_C + col],    w2 = jW2[d * A3_C + col];
    v8f acc = nt ? acc1 : acc0;
    #pragma unroll
    for (int j = 0; j < 8; ++j) {
      int row = j + 8 * half;
      float h = acc[j] + bb;
      h = h * (BN_INVC * g_) + be;
      h = (h > 0.f) ? h : (__expf(h) - 1.f);    // ELU
      hbuf[row][col] = h * w2;
    }
  }
  __syncthreads();
  int r = threadIdx.x >> 4, cc = threadIdx.x & 15;
  float s = 0.f;
  #pragma unroll
  for (int m = 0; m < 16; ++m) s += hbuf[r][cc + m * 16];
  rbuf[r][cc] = s;
  __syncthreads();
  if (threadIdx.x < 16) {
    float s2 = 0.f;
    #pragma unroll
    for (int m = 0; m < 16; ++m) s2 += rbuf[threadIdx.x][m];
    int t = tt * 16 + threadIdx.x;
    xfina[(long)t * DIME_C + d] = s2 + jb2[d];
  }
}

// ---------------------------------------------------------------------------
// Stage 4: GRU decode. Block = 32 rows, 4 waves, all 24 steps.
// h kept in f32 registers + bf16 A-fragment LDS copy; whh^T / cW1^T B-frags
// stream from L2 (pre-swizzled). Gates fused in VALU.
// ---------------------------------------------------------------------------
__global__ void __launch_bounds__(128)
gru_kernel(const float* __restrict__ xfina,
           const __bf16* __restrict__ bwhh,  // [2][24][4][512]
           const __bf16* __restrict__ bcw1,  // [2][8][4][512]
           const float* __restrict__ wih0, const float* __restrict__ bih0,
           const float* __restrict__ bhh0, const float* __restrict__ cb10,
           const float* __restrict__ cW20, const float* __restrict__ cb20,
           const float* __restrict__ wih1, const float* __restrict__ bih1,
           const float* __restrict__ bhh1, const float* __restrict__ cb11,
           const float* __restrict__ cW21, const float* __restrict__ cb21,
           float* __restrict__ out) {
  __shared__ __attribute__((aligned(32))) __bf16 hsf[2 * 4 * 512];  // A-frag layout [mt][kt][lane][i]
  __shared__ float wihL[384], bihL[384], bhhL[384], cb1L[HDIM], cW2L[HDIM];
  __shared__ float xinL[32], outbuf[32];

  int tag = blockIdx.y;
  const float* wih = tag ? wih1 : wih0;
  const float* bih = tag ? bih1 : bih0;
  const float* bhh = tag ? bhh1 : bhh0;
  const float* cb1 = tag ? cb11 : cb10;
  const float* cW2 = tag ? cW21 : cW20;
  float cb2v = (tag ? cb21 : cb20)[0];

  int tid = threadIdx.x, lane = tid & 31, wv = tid >> 5;
  for (int i = tid; i < 384; i += 128) { wihL[i] = wih[i]; bihL[i] = bih[i]; bhhL[i] = bhh[i]; }
  if (tid < 128) { cb1L[tid] = cb1[tid]; cW2L[tid] = cW2[tid]; }
  for (int i = tid; i < 4096; i += 128) hsf[i] = (__bf16)0.0f;
  long rowbase = (long)blockIdx.x * 32;
  if (tid < 32) xinL[tid] = xfina[rowbase + tid];
  __syncthreads();

  const __bf16* bwh = bwhh + (long)tag * 24 * 4 * 512;
  const __bf16* bcw = bcw1 + (long)tag * 8 * 4 * 512;
  float* outbase = out + (long)tag * T_DIM * PL_C * DIME_C;

  float hstate[2][2][8];
  #pragma unroll
  for (int mt = 0; mt < 2; ++mt)
    #pragma unroll
    for (int nt2 = 0; nt2 < 2; ++nt2)
      #pragma unroll
      for (int j = 0; j < 8; ++j) hstate[mt][nt2][j] = 0.f;

  int half = lane >> 4, ln = lane & 15;
  const v8f zero8 = {0.f,0.f,0.f,0.f,0.f,0.f,0.f,0.f};

  for (int p = 0; p < PL_C; ++p) {
    // ---- gh = h @ whh^T : 3 gate sections x 2 ntiles per wave x 2 mtiles
    v8f acc[3][2][2];
    #pragma unroll
    for (int g = 0; g < 3; ++g)
      #pragma unroll
      for (int nt2 = 0; nt2 < 2; ++nt2) { acc[g][nt2][0] = zero8; acc[g][nt2][1] = zero8; }
    #pragma unroll
    for (int kt = 0; kt < 4; ++kt) {
      v16bf a0 = *(const v16bf*)(hsf + ((0 * 4 + kt) * 32 + lane) * 16);
      v16bf a1 = *(const v16bf*)(hsf + ((1 * 4 + kt) * 32 + lane) * 16);
      #pragma unroll
      for (int g = 0; g < 3; ++g) {
        #pragma unroll
        for (int nt2 = 0; nt2 < 2; ++nt2) {
          int ntile = g * 8 + wv * 2 + nt2;
          v16bf b = *(const v16bf*)(bwh + (((long)ntile * 4 + kt) * 32 + lane) * 16);
          acc[g][nt2][0] = __builtin_amdgcn_wmma_f32_16x16x32_bf16(false, a0, false, b, (short)0, acc[g][nt2][0], false, false);
          acc[g][nt2][1] = __builtin_amdgcn_wmma_f32_16x16x32_bf16(false, a1, false, b, (short)0, acc[g][nt2][1], false, false);
        }
      }
    }
    __syncthreads();   // all reads of previous h in hsf complete

    // ---- gates + hn (fused gi outer product), write hn back swizzled
    #pragma unroll
    for (int mt = 0; mt < 2; ++mt) {
      #pragma unroll
      for (int nt2 = 0; nt2 < 2; ++nt2) {
        int col = wv * 32 + nt2 * 16 + ln;        // 0..127
        float wr = wihL[col],        br = bihL[col],        hrb = bhhL[col];
        float wz = wihL[128 + col],  bz = bihL[128 + col],  hzb = bhhL[128 + col];
        float wg = wihL[256 + col],  bg = bihL[256 + col],  hgb = bhhL[256 + col];
        int kt2 = col >> 5, kk = col & 31;
        int h2 = (kk >> 3) & 1;
        int i2 = (kk & 7) | ((kk & 16) >> 1);
        #pragma unroll
        for (int j = 0; j < 8; ++j) {
          int row = mt * 16 + j + 8 * half;
          float xi = xinL[row];
          float sr = acc[0][nt2][mt][j] + hrb + xi * wr + br;
          float sz = acc[1][nt2][mt][j] + hzb + xi * wz + bz;
          float r = 1.f / (1.f + __expf(-sr));
          float z = 1.f / (1.f + __expf(-sz));
          float hg = acc[2][nt2][mt][j] + hgb;
          float gg = tanhf(xi * wg + bg + r * hg);
          float hn = (1.f - z) * gg + z * hstate[mt][nt2][j];
          hstate[mt][nt2][j] = hn;
          int lane_t = (row & 15) + 16 * h2;
          hsf[(((row >> 4) * 4 + kt2) * 32 + lane_t) * 16 + i2] = (__bf16)hn;
        }
      }
    }
    if (tid < 32) outbuf[tid] = 0.f;
    __syncthreads();   // hn visible; outbuf zeroed

    // ---- mid = elu(hn @ cW1^T + cb1); out = mid @ cW2 + cb2
    v8f mm[2][2];
    mm[0][0] = zero8; mm[0][1] = zero8; mm[1][0] = zero8; mm[1][1] = zero8;
    #pragma unroll
    for (int kt = 0; kt < 4; ++kt) {
      v16bf a0 = *(const v16bf*)(hsf + ((0 * 4 + kt) * 32 + lane) * 16);
      v16bf a1 = *(const v16bf*)(hsf + ((1 * 4 + kt) * 32 + lane) * 16);
      #pragma unroll
      for (int nt2 = 0; nt2 < 2; ++nt2) {
        int ntile = wv * 2 + nt2;
        v16bf b = *(const v16bf*)(bcw + (((long)ntile * 4 + kt) * 32 + lane) * 16);
        mm[nt2][0] = __builtin_amdgcn_wmma_f32_16x16x32_bf16(false, a0, false, b, (short)0, mm[nt2][0], false, false);
        mm[nt2][1] = __builtin_amdgcn_wmma_f32_16x16x32_bf16(false, a1, false, b, (short)0, mm[nt2][1], false, false);
      }
    }
    #pragma unroll
    for (int mt = 0; mt < 2; ++mt) {
      #pragma unroll
      for (int nt2 = 0; nt2 < 2; ++nt2) {
        int col = wv * 32 + nt2 * 16 + ln;
        float cb = cb1L[col], w2 = cW2L[col];
        #pragma unroll
        for (int j = 0; j < 8; ++j) {
          int row = mt * 16 + j + 8 * half;
          float m = mm[nt2][mt][j] + cb;
          m = (m > 0.f) ? m : (__expf(m) - 1.f);
          atomicAdd(&outbuf[row], m * w2);
        }
      }
    }
    __syncthreads();   // reductions done
    if (tid < 32) {
      float o = outbuf[tid] + cb2v;
      long rg = rowbase + tid;
      int t = (int)(rg >> 3), dch = (int)(rg & 7);
      outbase[((long)t * PL_C + p) * DIME_C + dch] = o;
      xinL[tid] = o;
    }
    __syncthreads();   // xin visible for next step
  }
}

// ---------------------------------------------------------------------------
extern "C" void kernel_launch(void* const* d_in, const int* in_sizes, int n_in,
                              void* d_out, int out_size, void* d_ws, size_t ws_size,
                              hipStream_t stream) {
  const float* x_mul   = (const float*)d_in[0];
  const float* merge_w = (const float*)d_in[1];
  const float* linear_w = (const float*)d_in[2];
  const float* linear_b = (const float*)d_in[3];
  const float* jW1 = (const float*)d_in[4];
  const float* jb1 = (const float*)d_in[5];
  const float* jgamma = (const float*)d_in[6];
  const float* jbeta  = (const float*)d_in[7];
  const float* jW2 = (const float*)d_in[8];
  const float* jb2 = (const float*)d_in[9];
  const float* wih0 = (const float*)d_in[10];
  const float* whh0 = (const float*)d_in[11];
  const float* bih0 = (const float*)d_in[12];
  const float* bhh0 = (const float*)d_in[13];
  const float* cW1_0 = (const float*)d_in[14];
  const float* cb1_0 = (const float*)d_in[15];
  const float* cW2_0 = (const float*)d_in[16];
  const float* cb2_0 = (const float*)d_in[17];
  const float* wih1 = (const float*)d_in[18];
  const float* whh1 = (const float*)d_in[19];
  const float* bih1 = (const float*)d_in[20];
  const float* bhh1 = (const float*)d_in[21];
  const float* cW1_1 = (const float*)d_in[22];
  const float* cb1_1 = (const float*)d_in[23];
  const float* cW2_1 = (const float*)d_in[24];
  const float* cb2_1 = (const float*)d_in[25];
  (void)in_sizes; (void)n_in; (void)out_size; (void)ws_size;

  // workspace layout
  char* ws = (char*)d_ws;
  size_t off = 0;
  float*  xw     = (float*)(ws + off);  off += (size_t)T_DIM * A3_C * 4;          // 8 MB
  __bf16* afrag  = (__bf16*)(ws + off); off += (size_t)512 * 8 * 512 * 2;         // 4 MB
  __bf16* bfragJ = (__bf16*)(ws + off); off += (size_t)8 * 16 * 8 * 512 * 2;      // 1 MB
  __bf16* bwhh   = (__bf16*)(ws + off); off += (size_t)2 * 24 * 4 * 512 * 2;      // 192 KB
  __bf16* bcw1   = (__bf16*)(ws + off); off += (size_t)2 * 8 * 4 * 512 * 2;       // 64 KB
  float*  xfina  = (float*)(ws + off);  off += (size_t)T_DIM * DIME_C * 4;        // 256 KB

  const long TPD = (long)T_DIM * PL_C * DIME_C;  // 1,572,864
  float* pred0 = (float*)d_out;
  float* rep_o = (float*)d_out + 2 * TPD;

  // --- weight pre-swizzle (tiny) ---
  swizzleB_kernel<<<dim3(256, 8), 256, 0, stream>>>(jW1, bfragJ, 16, 8, 256, 256, 0,
                                                    (long)256 * 256, (long)16 * 8 * 512);
  swizzleB_kernel<<<dim3(192, 1), 256, 0, stream>>>(whh0, bwhh, 24, 4, 384, 128, 1, 0, 0);
  swizzleB_kernel<<<dim3(192, 1), 256, 0, stream>>>(whh1, bwhh + (long)24 * 4 * 512, 24, 4, 384, 128, 1, 0, 0);
  swizzleB_kernel<<<dim3(64, 1), 256, 0, stream>>>(cW1_0, bcw1, 8, 4, 128, 128, 1, 0, 0);
  swizzleB_kernel<<<dim3(64, 1), 256, 0, stream>>>(cW1_1, bcw1 + (long)8 * 4 * 512, 8, 4, 128, 128, 1, 0, 0);

  // --- stage 1: channel merge (memory bound, 256 MB, b128 vectorized) ---
  merge_kernel<<<dim3((T_DIM * A3_C) / (256 * 4)), 256, 0, stream>>>(x_mul, merge_w, xw);

  // --- stage 2: window + L2 norm -> rep (f32) + swizzled bf16 A fragments ---
  window_norm_kernel<<<dim3(T_DIM), 256, 0, stream>>>(xw, linear_w, linear_b, rep_o, afrag);

  // --- stage 3: jiang MLP (WMMA) -> x_fina ---
  jiang_kernel<<<dim3(512, 8), 256, 0, stream>>>(afrag, bfragJ, jb1, jgamma, jbeta, jW2, jb2, xfina);

  // --- stage 4: dual GRU decode (WMMA, dominant compute) ---
  gru_kernel<<<dim3(2048, 2), 128, 0, stream>>>(xfina, bwhh, bcw1,
                                                wih0, bih0, bhh0, cb1_0, cW2_0, cb2_0,
                                                wih1, bih1, bhh1, cb1_1, cW2_1, cb2_1,
                                                pred0);
}
